// LocalMHA_43078521978895
// MI455X (gfx1250) — compile-verified, matrix operand order checked by reading, not run on hardware
//
#include <hip/hip_runtime.h>
#include <hip/hip_bf16.h>

typedef __attribute__((ext_vector_type(16))) _Float16 v16h;
typedef __attribute__((ext_vector_type(8)))  float    v8f;
typedef __attribute__((ext_vector_type(4)))  float    f4_t;
typedef __attribute__((ext_vector_type(4)))  int      i4_t;

#if defined(__AMDGCN__) &&                                                \
    __has_builtin(__builtin_amdgcn_global_load_async_to_lds_b128) &&      \
    __has_builtin(__builtin_amdgcn_s_wait_asynccnt)
#define USE_ASYNC 1
// global (AS1) and LDS (AS3) int4 pointers, as required by the async builtin
#define GPTR_B128(p) ((__attribute__((address_space(1))) i4_t*)(p))
#define LPTR_B128(p) ((__attribute__((address_space(3))) i4_t*)(p))
#else
#define USE_ASYNC 0
#endif

// ---------------------------------------------------------------------------
// Fragment loader: 16-bit A/B fragment for V_WMMA_F32_16X16X32_F16.
// Lane l (wave32): row/col = l%16, K halves at 8*(l/16)+[0..7] and +16.
// p points at (row base + (l/16)*8) halfs; 16-byte aligned.
// ---------------------------------------------------------------------------
__device__ __forceinline__ v16h frag_ld(const _Float16* p) {
    union { f4_t f[2]; v16h v; } u;
    u.f[0] = *(const f4_t*)(p);
    u.f[1] = *(const f4_t*)(p + 16);
    return u.v;
}

__device__ __forceinline__ v8f wmma_f16(v16h a, v16h b, v8f c) {
    return __builtin_amdgcn_wmma_f32_16x16x32_f16(false, a, false, b, (short)0, c,
                                                  false, false);
}

// ---------------------------------------------------------------------------
// f32 -> f16 conversion
// ---------------------------------------------------------------------------
__global__ void f32_to_f16_kernel(const float* __restrict__ in,
                                  _Float16* __restrict__ out, int n) {
    for (int i = blockIdx.x * blockDim.x + threadIdx.x; i < n;
         i += gridDim.x * blockDim.x)
        out[i] = (_Float16)in[i];
}

// ---------------------------------------------------------------------------
// Transposing f32 -> f16: in R x C (row-major) -> out C x R (row-major).
// 32x32 LDS tile, 256 threads.
// ---------------------------------------------------------------------------
__global__ __launch_bounds__(256) void transpose_f32_to_f16_kernel(
    const float* __restrict__ in, _Float16* __restrict__ out, int R, int C) {
    __shared__ float tile[32][33];
    const int c0 = blockIdx.x * 32, r0 = blockIdx.y * 32;
    const int tx = threadIdx.x & 31, ty = threadIdx.x >> 5;  // 32 x 8
#pragma unroll
    for (int dy = 0; dy < 32; dy += 8)
        tile[ty + dy][tx] = in[(size_t)(r0 + ty + dy) * C + c0 + tx];
    __syncthreads();
#pragma unroll
    for (int dy = 0; dy < 32; dy += 8)
        out[(size_t)(c0 + ty + dy) * R + r0 + tx] = (_Float16)tile[tx][ty + dy];
}

// ---------------------------------------------------------------------------
// Tiled WMMA GEMM: C[MxN] = A[MxK] * Bt[NxK]^T  (B pre-transposed to N x K).
// Block tile 128x128, BK=32, 256 threads = 8 waves (4 along M x 2 along N),
// each wave owns a 32x64 output patch = 2x4 WMMA tiles.
// Double-buffered LDS staging via GLOBAL_LOAD_ASYNC_TO_LDS_B128 (ASYNCcnt)
// when available; synchronous b128 staging otherwise.
// ---------------------------------------------------------------------------
template <typename OutT>
__global__ __launch_bounds__(256) void gemm_wmma_kernel(
    const _Float16* __restrict__ A, const _Float16* __restrict__ Bt,
    OutT* __restrict__ C, int M, int N, int K) {
    __shared__ _Float16 As[2][128][48];  // [buf][m][k], 96B row stride
    __shared__ _Float16 Bs[2][128][48];  // [buf][n][k]

    const int tid   = threadIdx.x;
    const int lane  = tid & 31;
    const int wid   = tid >> 5;
    const int mrel  = lane & 15;
    const int khalf = lane >> 4;
    const int wm    = wid & 3;   // 32-row band
    const int wn    = wid >> 2;  // 64-col band
    const int bM    = blockIdx.y * 128;
    const int bN    = blockIdx.x * 128;

    // Tile loader mapping: thread -> (row 0..63 [+64], 8-half column chunk)
    const int ldr = tid >> 2;
    const int ldc = (tid & 3) << 3;
    const _Float16* Ab = A + (size_t)(bM + ldr) * K + ldc;
    const _Float16* Bb = Bt + (size_t)(bN + ldr) * K + ldc;
    const size_t rowskip = (size_t)64 * K;

    v8f acc[2][4] = {};
    const int nIter = K >> 5;

#if USE_ASYNC
    auto prefetch = [&](int it, int buf) {
        const int k0 = it << 5;
#pragma unroll
        for (int p = 0; p < 2; ++p) {
            __builtin_amdgcn_global_load_async_to_lds_b128(
                GPTR_B128(Ab + p * rowskip + k0),
                LPTR_B128(&As[buf][ldr + p * 64][ldc]), 0, 0);
            __builtin_amdgcn_global_load_async_to_lds_b128(
                GPTR_B128(Bb + p * rowskip + k0),
                LPTR_B128(&Bs[buf][ldr + p * 64][ldc]), 0, 0);
        }
    };
    prefetch(0, 0);
    for (int it = 0; it < nIter; ++it) {
        const int buf = it & 1;
        __syncthreads();  // previous readers of buf^1 are done
        if (it + 1 < nIter) {
            prefetch(it + 1, buf ^ 1);
            __builtin_amdgcn_s_wait_asynccnt(4);  // tile `it` (4 older ops) landed
        } else {
            __builtin_amdgcn_s_wait_asynccnt(0);
        }
        __syncthreads();  // tile `it` visible to every wave

        v16h af[2], bf[4];
#pragma unroll
        for (int ti = 0; ti < 2; ++ti)
            af[ti] = frag_ld(&As[buf][wm * 32 + ti * 16 + mrel][khalf * 8]);
#pragma unroll
        for (int tj = 0; tj < 4; ++tj)
            bf[tj] = frag_ld(&Bs[buf][wn * 64 + tj * 16 + mrel][khalf * 8]);
#pragma unroll
        for (int ti = 0; ti < 2; ++ti)
#pragma unroll
            for (int tj = 0; tj < 4; ++tj)
                acc[ti][tj] = wmma_f16(af[ti], bf[tj], acc[ti][tj]);
    }
#else
    for (int it = 0; it < nIter; ++it) {
        const int k0 = it << 5;
#pragma unroll
        for (int p = 0; p < 2; ++p) {
            *(f4_t*)&As[0][ldr + p * 64][ldc] = *(const f4_t*)(Ab + p * rowskip + k0);
            *(f4_t*)&Bs[0][ldr + p * 64][ldc] = *(const f4_t*)(Bb + p * rowskip + k0);
        }
        __syncthreads();
        v16h af[2], bf[4];
#pragma unroll
        for (int ti = 0; ti < 2; ++ti)
            af[ti] = frag_ld(&As[0][wm * 32 + ti * 16 + mrel][khalf * 8]);
#pragma unroll
        for (int tj = 0; tj < 4; ++tj)
            bf[tj] = frag_ld(&Bs[0][wn * 64 + tj * 16 + mrel][khalf * 8]);
#pragma unroll
        for (int ti = 0; ti < 2; ++ti)
#pragma unroll
            for (int tj = 0; tj < 4; ++tj)
                acc[ti][tj] = wmma_f16(af[ti], bf[tj], acc[ti][tj]);
        __syncthreads();
    }
#endif

    // C/D layout: lane l, vgpr r -> row = 8*(l/16)+r, col = l%16 of each tile.
#pragma unroll
    for (int ti = 0; ti < 2; ++ti)
#pragma unroll
        for (int tj = 0; tj < 4; ++tj)
#pragma unroll
            for (int r = 0; r < 8; ++r) {
                int row = bM + wm * 32 + ti * 16 + khalf * 8 + r;
                int col = bN + wn * 64 + tj * 16 + mrel;
                C[(size_t)row * N + col] = (OutT)acc[ti][tj][r];
            }
}

// ---------------------------------------------------------------------------
// Local windowed attention, one workgroup per (b, head, window).
// ---------------------------------------------------------------------------
__global__ __launch_bounds__(256) void local_attn_kernel(
    const _Float16* __restrict__ qkv,  // (4, 4096, 3072) f16
    _Float16* __restrict__ attn_out) { // (4, 4096, 1024) f16
    // Aliased LDS (101,376 bytes):
    //   phase 1: Qs = smem[0 .. 18432)       128 x 72 halfs
    //            Ks = smem[18432 .. 55296)   256 x 72 halfs
    //   phase 2: At = smem[0 .. 67584)       128 x 264 halfs (overlays Qs+Ks)
    //   always : Vs = smem[67584 .. 101376)   64 x 264 halfs ([d][j])
    __shared__ __align__(16) char smem[101376];
    _Float16* Qs = (_Float16*)(smem);
    _Float16* Ks = (_Float16*)(smem + 18432);
    _Float16* At = (_Float16*)(smem);
    _Float16* Vs = (_Float16*)(smem + 67584);

    const int tid   = threadIdx.x;
    const int lane  = tid & 31;
    const int wid   = tid >> 5;
    const int mrel  = lane & 15;
    const int khalf = lane >> 4;
    const int qbase = wid * 16;

    const int blk = blockIdx.x;
    const int wi  = blk & 31;
    const int hh  = (blk >> 5) & 15;
    const int bb  = blk >> 9;

    const _Float16* qg = qkv + ((size_t)(bb * 4096 + wi * 128)) * 3072 + hh * 64;
    const _Float16* kg = qkv + ((size_t)bb * 4096) * 3072 + 1024 + hh * 64;
    const _Float16* vg = qkv + ((size_t)bb * 4096) * 3072 + 2048 + hh * 64;

    const float inv32 = 1.0f / 32.0f;

    // Q: RoPE at position 128+i, scaled dh^-0.5
    for (int idx = tid; idx < 128 * 32; idx += 256) {
        int i = idx >> 5, d = idx & 31;
        float q0 = (float)qg[(size_t)i * 3072 + d];
        float q1 = (float)qg[(size_t)i * 3072 + d + 32];
        float ang = (float)(128 + i) * __powf(10000.0f, -(float)d * inv32);
        float s, c;
        __sincosf(ang, &s, &c);
        Qs[i * 72 + d]      = (_Float16)((q0 * c - q1 * s) * 0.125f);
        Qs[i * 72 + d + 32] = (_Float16)((q1 * c + q0 * s) * 0.125f);
    }
    // K: look_around pad -1, RoPE at position j
    for (int idx = tid; idx < 256 * 32; idx += 256) {
        int j = idx >> 5, d = idx & 31;
        int n = wi * 128 - 128 + j;
        float k0 = -1.0f, k1 = -1.0f;
        if (n >= 0) {
            k0 = (float)kg[(size_t)n * 3072 + d];
            k1 = (float)kg[(size_t)n * 3072 + d + 32];
        }
        float ang = (float)j * __powf(10000.0f, -(float)d * inv32);
        float s, c;
        __sincosf(ang, &s, &c);
        Ks[j * 72 + d]      = (_Float16)(k0 * c - k1 * s);
        Ks[j * 72 + d + 32] = (_Float16)(k1 * c + k0 * s);
    }
    // V: pad -1, stored transposed [d][j]
    for (int idx = tid; idx < 256 * 64; idx += 256) {
        int j = idx >> 6, d = idx & 63;
        int n = wi * 128 - 128 + j;
        Vs[d * 264 + j] = (n >= 0) ? vg[(size_t)n * 3072 + d] : (_Float16)(-1.0f);
    }
    __syncthreads();

    // sim = q k^T : 16 j-tiles x 2 K-steps
    v8f sim[16] = {};
    {
        v16h aq0 = frag_ld(&Qs[(qbase + mrel) * 72 + khalf * 8]);
        v16h aq1 = frag_ld(&Qs[(qbase + mrel) * 72 + 32 + khalf * 8]);
#pragma unroll
        for (int t = 0; t < 16; ++t) {
            v16h bk0 = frag_ld(&Ks[(t * 16 + mrel) * 72 + khalf * 8]);
            v16h bk1 = frag_ld(&Ks[(t * 16 + mrel) * 72 + 32 + khalf * 8]);
            sim[t] = wmma_f16(aq0, bk0, sim[t]);
            sim[t] = wmma_f16(aq1, bk1, sim[t]);
        }
    }

    // mask + softmax in C-layout registers
#pragma unroll
    for (int t = 0; t < 16; ++t)
#pragma unroll
        for (int r = 0; r < 8; ++r) {
            int i = qbase + khalf * 8 + r;
            int j = t * 16 + mrel;
            bool ok = (j >= i) && (j <= i + 128) && !(wi == 0 && j < 128);
            if (!ok) sim[t][r] = -3.0e38f;
        }
#pragma unroll
    for (int r = 0; r < 8; ++r) {
        float mx = -3.0e38f;
#pragma unroll
        for (int t = 0; t < 16; ++t) mx = fmaxf(mx, sim[t][r]);
        for (int o = 8; o > 0; o >>= 1) mx = fmaxf(mx, __shfl_xor(mx, o, 32));
        float sum = 0.0f;
#pragma unroll
        for (int t = 0; t < 16; ++t) {
            float e = __expf(sim[t][r] - mx);
            sim[t][r] = e;
            sum += e;
        }
        for (int o = 8; o > 0; o >>= 1) sum += __shfl_xor(sum, o, 32);
        float inv = 1.0f / sum;
#pragma unroll
        for (int t = 0; t < 16; ++t) sim[t][r] *= inv;
    }

    __syncthreads();  // everyone done with Qs/Ks before At overlays them
#pragma unroll
    for (int r = 0; r < 8; ++r) {
        int i = qbase + khalf * 8 + r;
#pragma unroll
        for (int t = 0; t < 16; ++t)
            At[i * 264 + t * 16 + mrel] = (_Float16)sim[t][r];
    }
    __syncthreads();

    // out = attn @ v : 4 d-tiles, 8 K-steps over j
    v8f o[4] = {};
#pragma unroll
    for (int s = 0; s < 8; ++s) {
        v16h aa = frag_ld(&At[(qbase + mrel) * 264 + s * 32 + khalf * 8]);
#pragma unroll
        for (int dt = 0; dt < 4; ++dt) {
            v16h bv = frag_ld(&Vs[(dt * 16 + mrel) * 264 + s * 32 + khalf * 8]);
            o[dt] = wmma_f16(aa, bv, o[dt]);
        }
    }

    _Float16* op = attn_out + ((size_t)(bb * 4096 + wi * 128)) * 1024 + hh * 64;
#pragma unroll
    for (int dt = 0; dt < 4; ++dt)
#pragma unroll
        for (int r = 0; r < 8; ++r) {
            int i = qbase + khalf * 8 + r;
            op[(size_t)i * 1024 + dt * 16 + mrel] = (_Float16)o[dt][r];
        }
}

// ---------------------------------------------------------------------------
// Launch
// ---------------------------------------------------------------------------
extern "C" void kernel_launch(void* const* d_in, const int* in_sizes, int n_in,
                              void* d_out, int out_size, void* d_ws,
                              size_t ws_size, hipStream_t stream) {
    const float* x     = (const float*)d_in[0];  // (4,4096,1024)
    const float* w_qkv = (const float*)d_in[1];  // (1024,3072)
    const float* w_out = (const float*)d_in[2];  // (1024,1024)
    float* out = (float*)d_out;                  // (4,4096,1024)

    char* ws = (char*)d_ws;
    _Float16* xh     = (_Float16*)(ws);                     // 33,554,432 B
    _Float16* wqkvT  = (_Float16*)(ws + (size_t)33554432);  //  6,291,456 B (3072x1024)
    _Float16* woutT  = (_Float16*)(ws + (size_t)39845888);  //  2,097,152 B (1024x1024)
    _Float16* qkvh   = (_Float16*)(ws + (size_t)41943040);  // 100,663,296 B
    _Float16* atth   = (_Float16*)(ws + (size_t)142606336); // 33,554,432 B

    f32_to_f16_kernel<<<2048, 256, 0, stream>>>(x, xh, 16777216);
    transpose_f32_to_f16_kernel<<<dim3(96, 32), 256, 0, stream>>>(w_qkv, wqkvT,
                                                                  1024, 3072);
    transpose_f32_to_f16_kernel<<<dim3(32, 32), 256, 0, stream>>>(w_out, woutT,
                                                                  1024, 1024);

    // qkv = x @ W_qkv  (16384 x 1024 x 3072)
    gemm_wmma_kernel<_Float16>
        <<<dim3(24, 128), 256, 0, stream>>>(xh, wqkvT, qkvh, 16384, 3072, 1024);

    // windowed attention: 4 * 16 * 32 = 2048 blocks
    local_attn_kernel<<<2048, 256, 0, stream>>>(qkvh, atth);

    // out = attn_out @ W_out  (16384 x 1024 x 1024), f32 result
    gemm_wmma_kernel<float>
        <<<dim3(8, 128), 256, 0, stream>>>(atth, woutT, out, 16384, 1024, 1024);
}
